// CustomRNN_59528246723083
// MI455X (gfx1250) — matrix operand order repaired and checked by reading, not compiled, not run on hardware
//
#include <hip/hip_runtime.h>
#include <hip/hip_bf16.h>

typedef __bf16 bf16_t;
typedef __attribute__((ext_vector_type(16))) __bf16 v16bf;
typedef __attribute__((ext_vector_type(8)))  __bf16 v8bf;
typedef __attribute__((ext_vector_type(8)))  float  v8f;

constexpr int T0 = 512, B0 = 64, D0 = 1024, H0 = 1024;
constexpr int KTOT = D0 + H0;        // 2048 fused K (input + recurrent)
constexpr int NG   = 4 * H0;         // 4096 gate columns

// ---------- helpers ----------

__device__ __forceinline__ float fsigmoid(float x) {
    return __builtin_amdgcn_rcpf(1.0f + __expf(-x));
}
__device__ __forceinline__ float ftanh(float x) {
    return 2.0f * __builtin_amdgcn_rcpf(1.0f + __expf(-2.0f * x)) - 1.0f;
}

// A fragment: 16x32 bf16 tile, row-major source, lda = 1024 elements.
// ISA 7.12.2: lanes 0-15 row M=lane, K chunks [0..7],[16..23];
// lanes 16-31 same rows, K chunks [8..15],[24..31].
__device__ __forceinline__ v16bf load_a16x32(const bf16_t* p) {
    union { v16bf v; v8bf h[2]; } u;
    u.h[0] = *(const v8bf*)(p);
    u.h[1] = *(const v8bf*)(p + 16);
    return u.v;
}

// B fragment: 32x16 (KxN) bf16 tile; weights stored K-contiguous per column
// (row-major [N][K], ldb = KTOT). Lanes 0-15: col N=lane, K chunks [0..7],[16..23];
// lanes 16-31: col N=lane-16, K chunks [8..15],[24..31].
__device__ __forceinline__ v16bf load_b32x16(const bf16_t* p) {
    union { v16bf v; v8bf h[2]; } u;
    u.h[0] = *(const v8bf*)(p);
    u.h[1] = *(const v8bf*)(p + 16);
    return u.v;
}

#define WMMA_BF16(ACC, A, B) \
    ACC = __builtin_amdgcn_wmma_f32_16x16x32_bf16(false, A, false, B, (short)0, ACC, false, false)

// ---------- one-time prep kernels ----------

__global__ void convert_x_kernel(const float* __restrict__ x, bf16_t* __restrict__ xbf,
                                 size_t n) {
    size_t i = (size_t)blockIdx.x * blockDim.x + threadIdx.x;
    if (i < n) xbf[i] = (bf16_t)x[i];
}

// Build Wcomb[dir][n][k] (bf16, K-contiguous): k<D0 from Wih[k][n], else Whh[k-D0][n].
__global__ void pack_w_kernel(const float* __restrict__ Wih,  const float* __restrict__ Whh,
                              const float* __restrict__ WihR, const float* __restrict__ WhhR,
                              bf16_t* __restrict__ Wcomb) {
    size_t idx = (size_t)blockIdx.x * blockDim.x + threadIdx.x;
    const size_t per_dir = (size_t)NG * KTOT;           // 2^23
    if (idx >= 2 * per_dir) return;
    const int dir = (int)(idx >> 23);
    const size_t r = idx & (per_dir - 1);
    const int n = (int)(r >> 11);                       // KTOT = 2048
    const int k = (int)(r & (KTOT - 1));
    const float* wi = dir ? WihR : Wih;
    const float* wh = dir ? WhhR : Whh;
    float v = (k < D0) ? wi[(size_t)k * NG + n] : wh[(size_t)(k - D0) * NG + n];
    Wcomb[idx] = (bf16_t)v;
}

__global__ void init_state_kernel(bf16_t* __restrict__ h0, float* __restrict__ c0, int n) {
    int i = blockIdx.x * blockDim.x + threadIdx.x;
    if (i < n) { h0[i] = (bf16_t)0.0f; c0[i] = 0.0f; }
}

// ---------- recurrent step ----------
// grid = (H/64, 2 dirs), block = 256 (8 waves).
// Wave w: gate = w&3, batch half m0 = (w>>2)*32. Each wave: 2 (M) x 4 (N) WMMA tiles.
__global__ __launch_bounds__(256) void lstm_step_kernel(
    const bf16_t* __restrict__ xbf,      // [T,B,D] bf16
    const bf16_t* __restrict__ Wcomb,    // [2][NG][KTOT] bf16
    const float*  __restrict__ bias_f,   // [NG]
    const float*  __restrict__ bias_r,   // [NG]
    const bf16_t* __restrict__ h_prev,   // [2][B][H] bf16
    bf16_t*       __restrict__ h_next,   // [2][B][H] bf16
    float*        __restrict__ c_state,  // [2][B][H] f32 (in/out)
    float*        __restrict__ out,      // concat(output, hT, cT)
    int s) {
    __shared__ float glds[4][64][64];    // 64 KB

    const int dir  = blockIdx.y;
    const int t_x  = dir ? (T0 - 1 - s) : s;
    const int n0   = blockIdx.x * 64;
    const int wave = threadIdx.x >> 5;
    const int lane = threadIdx.x & 31;
    const int gate = wave & 3;
    const int m0   = (wave >> 2) * 32;
    const int colbase = gate * H0 + n0;

    // Per-lane fragment offsets (constant across K loop).
    const int am = lane & 15;                  // A row within tile
    const int ak = (lane >> 4) << 3;           // A k sub-offset: 0 or 8
    const int bn = lane & 15;                  // B column within tile
    const int bk = (lane >> 4) << 3;           // B k sub-offset: 0 or 8

    v8f acc[2][4];
    for (int mi = 0; mi < 2; ++mi)
        for (int ni = 0; ni < 4; ++ni)
            acc[mi][ni] = (v8f){0.f, 0.f, 0.f, 0.f, 0.f, 0.f, 0.f, 0.f};

    // Running pointers: pa0/pa1 at this lane's A element, pb at this lane's B element.
    const bf16_t* pb = Wcomb + (size_t)dir * NG * KTOT
                     + (size_t)(colbase + bn) * KTOT + bk;

    // --- K part 1: input projection (x_t @ Wih), k = 0..1023 ---
    {
        const bf16_t* xrow = xbf + (size_t)t_x * B0 * D0;
        const bf16_t* pa0 = xrow + (size_t)(m0 + am)      * D0 + ak;
        const bf16_t* pa1 = xrow + (size_t)(m0 + 16 + am) * D0 + ak;
        const bf16_t* pbk = pb;
        #pragma unroll 4
        for (int kb = 0; kb < D0 / 32; ++kb) {
            v16bf a0 = load_a16x32(pa0);
            v16bf a1 = load_a16x32(pa1);
            v16bf b0 = load_b32x16(pbk);
            v16bf b1 = load_b32x16(pbk + 16 * (size_t)KTOT);
            v16bf b2 = load_b32x16(pbk + 32 * (size_t)KTOT);
            v16bf b3 = load_b32x16(pbk + 48 * (size_t)KTOT);
            WMMA_BF16(acc[0][0], a0, b0);
            WMMA_BF16(acc[0][1], a0, b1);
            WMMA_BF16(acc[0][2], a0, b2);
            WMMA_BF16(acc[0][3], a0, b3);
            WMMA_BF16(acc[1][0], a1, b0);
            WMMA_BF16(acc[1][1], a1, b1);
            WMMA_BF16(acc[1][2], a1, b2);
            WMMA_BF16(acc[1][3], a1, b3);
            pa0 += 32; pa1 += 32; pbk += 32;
        }
    }
    // --- K part 2: recurrent projection (h_prev @ Whh), k = 1024..2047 ---
    {
        const bf16_t* hrow = h_prev + (size_t)dir * B0 * H0;
        const bf16_t* pa0 = hrow + (size_t)(m0 + am)      * H0 + ak;
        const bf16_t* pa1 = hrow + (size_t)(m0 + 16 + am) * H0 + ak;
        const bf16_t* pbk = pb + D0;
        #pragma unroll 4
        for (int kb = 0; kb < H0 / 32; ++kb) {
            v16bf a0 = load_a16x32(pa0);
            v16bf a1 = load_a16x32(pa1);
            v16bf b0 = load_b32x16(pbk);
            v16bf b1 = load_b32x16(pbk + 16 * (size_t)KTOT);
            v16bf b2 = load_b32x16(pbk + 32 * (size_t)KTOT);
            v16bf b3 = load_b32x16(pbk + 48 * (size_t)KTOT);
            WMMA_BF16(acc[0][0], a0, b0);
            WMMA_BF16(acc[0][1], a0, b1);
            WMMA_BF16(acc[0][2], a0, b2);
            WMMA_BF16(acc[0][3], a0, b3);
            WMMA_BF16(acc[1][0], a1, b0);
            WMMA_BF16(acc[1][1], a1, b1);
            WMMA_BF16(acc[1][2], a1, b2);
            WMMA_BF16(acc[1][3], a1, b3);
            pa0 += 32; pa1 += 32; pbk += 32;
        }
    }

    // Spill gate tiles to LDS (C layout: VGPR j -> row j + 8*(lane>=16), col = lane&15).
    {
        const float* bias = dir ? bias_r : bias_f;
        const int n    = lane & 15;
        const int mrow = (lane >> 4) << 3;
        for (int ni = 0; ni < 4; ++ni) {
            const float bv = bias[colbase + 16 * ni + n];
            for (int mi = 0; mi < 2; ++mi)
                for (int j = 0; j < 8; ++j)
                    glds[gate][m0 + 16 * mi + mrow + j][16 * ni + n] = acc[mi][ni][j] + bv;
        }
    }
    __syncthreads();

    // Fused LSTM cell epilogue: 256 threads cover 64x64 outputs, 16 each.
    {
        const int tid = threadIdx.x;
        const int m   = tid >> 2;            // batch row
        const int nb  = (tid & 3) * 16;
        float*  cptr = c_state + (size_t)dir * B0 * H0 + (size_t)m * H0 + n0;
        bf16_t* hn   = h_next  + (size_t)dir * B0 * H0 + (size_t)m * H0 + n0;
        float*  op   = out + ((size_t)t_x * B0 + m) * (2 * H0) + (size_t)dir * H0 + n0;
        const bool last_fwd = (dir == 0) && (s == T0 - 1);
        for (int i = 0; i < 16; ++i) {
            const int n = nb + i;
            const float ig = fsigmoid(glds[0][m][n]);
            const float fg = fsigmoid(glds[1][m][n]);
            const float gg = ftanh(glds[2][m][n]);
            const float og = fsigmoid(glds[3][m][n]);
            const float cp = cptr[n];
            const float cn = fg * cp + ig * gg;
            const float hv = og * ftanh(cn);
            cptr[n] = cn;
            hn[n]   = (bf16_t)hv;
            op[n]   = hv;
            if (last_fwd) {
                const size_t base = (size_t)T0 * B0 * 2 * H0 + (size_t)m * H0 + n0 + n;
                out[base] = hv;                          // h_T
                out[base + (size_t)B0 * H0] = cn;        // c_T
            }
        }
    }
}

// ---------- host ----------

extern "C" void kernel_launch(void* const* d_in, const int* in_sizes, int n_in,
                              void* d_out, int out_size, void* d_ws, size_t ws_size,
                              hipStream_t stream) {
    (void)in_sizes; (void)n_in; (void)out_size; (void)ws_size;
    const float* x     = (const float*)d_in[0];
    const float* Wih   = (const float*)d_in[1];
    const float* Whh   = (const float*)d_in[2];
    const float* b     = (const float*)d_in[3];
    const float* Wih_r = (const float*)d_in[4];
    const float* Whh_r = (const float*)d_in[5];
    const float* b_r   = (const float*)d_in[6];
    float* out = (float*)d_out;

    // workspace layout (256B aligned)
    char* ws = (char*)d_ws;
    size_t off = 0;
    auto take = [&](size_t bytes) { char* p = ws + off; off = (off + bytes + 255) & ~(size_t)255; return p; };
    bf16_t* xbf   = (bf16_t*)take((size_t)T0 * B0 * D0 * sizeof(bf16_t));   // 64 MB
    bf16_t* Wcomb = (bf16_t*)take((size_t)2 * NG * KTOT * sizeof(bf16_t));  // 32 MB
    bf16_t* hbuf  = (bf16_t*)take((size_t)2 * 2 * B0 * H0 * sizeof(bf16_t));// ping-pong x 2 dirs
    float*  cbuf  = (float*) take((size_t)2 * B0 * H0 * sizeof(float));

    // 1) convert x to bf16
    {
        size_t n = (size_t)T0 * B0 * D0;
        convert_x_kernel<<<dim3((unsigned)((n + 255) / 256)), dim3(256), 0, stream>>>(x, xbf, n);
    }
    // 2) pack + transpose weights: [2][NG][KTOT] bf16, K-contiguous
    {
        size_t n = (size_t)2 * NG * KTOT;
        pack_w_kernel<<<dim3((unsigned)((n + 255) / 256)), dim3(256), 0, stream>>>(Wih, Whh, Wih_r, Whh_r, Wcomb);
    }
    // 3) zero initial h (ping 0) and c for both directions
    {
        int n = 2 * B0 * H0;
        init_state_kernel<<<dim3((n + 255) / 256), dim3(256), 0, stream>>>(hbuf, cbuf, n);
    }
    // 4) 512 recurrent steps, both directions per launch
    const size_t hstride = (size_t)2 * B0 * H0;
    for (int s = 0; s < T0; ++s) {
        const bf16_t* h_prev = hbuf + (size_t)(s & 1) * hstride;
        bf16_t*       h_next = hbuf + (size_t)((s + 1) & 1) * hstride;
        lstm_step_kernel<<<dim3(H0 / 64, 2), dim3(256), 0, stream>>>(
            xbf, Wcomb, b, b_r, h_prev, h_next, cbuf, out, s);
    }
}
